// QuantizedLinear_46548855554492
// MI455X (gfx1250) — matrix operand order repaired and checked by reading, not compiled, not run on hardware
//
#include <hip/hip_runtime.h>
#include <hip/hip_bf16.h>
#include <stddef.h>

// ---------------------------------------------------------------------------
// QuantizedLinear for MI455X (gfx1250, wave32, WMMA)
//   y[m,n] = scale[n]*(x[m,:]·W8[n,:]) - scale[n]*zero[n]*rowsum(x[m,:]) + bias[n]
// f16 WMMA (int8 exact in f16), f32 accumulate. HBM traffic ~675MB (~29us at
// 23.3TB/s). 256x256 block tiles (512 thr, 16 waves, 64x64 per wave) keep
// L2->WGP traffic at K*(BM+BN)/block (11.5GB total). Double-buffered dynamic
// LDS (83KB of 320KB/WGP) with ONE barrier per K-step, b128 loads, short->f16
// weight dequant, rowsum(x) on the staging side, NT output stores to protect
// L2-resident tiles.
// ---------------------------------------------------------------------------

typedef __attribute__((ext_vector_type(16))) _Float16 v16h;
typedef __attribute__((ext_vector_type(8)))  _Float16 v8h;
typedef __attribute__((ext_vector_type(8)))  float    v8f;
typedef __attribute__((ext_vector_type(4)))  float    v4f;
typedef __attribute__((ext_vector_type(4)))  int      v4i;

#define BM 256
#define BN 256
#define BK 32
#define SAS 40            // LDS row stride in halves (32 data + 8 pad -> 80B)
#define TILEH (BM * SAS)  // halves per LDS buffer (10240)
#define KDIM 4096
#define NDIM 11008
#define MDIM 8192
#define NKT (KDIM / BK)   // 128 K-tiles
#define LDS_BYTES (4 * TILEH * 2 + BM * 4)   // sA[2]+sB[2] f16 + sRow f32

__device__ __forceinline__ void load_stage(const float* __restrict__ pA,
                                           const int* __restrict__ pW,
                                           v4f fa[4], v4i wa[4]) {
#pragma unroll
  for (int i = 0; i < 4; ++i) fa[i] = *(const v4f*)(pA + i * 4);
#pragma unroll
  for (int i = 0; i < 4; ++i) wa[i] = *(const v4i*)(pW + i * 4);
}

__device__ __forceinline__ void store_stage(_Float16* dA, _Float16* dW,
                                            const v4f fa[4], const v4i wa[4]) {
  v8h a0, a1, w0, w1;
#pragma unroll
  for (int i = 0; i < 8; ++i) {
    a0[i] = (_Float16)fa[i >> 2][i & 3];
    a1[i] = (_Float16)fa[2 + (i >> 2)][i & 3];
    // int8 weights: exact via i16 -> f16 (v_cvt_f16_i16), skips f32 hop
    w0[i] = (_Float16)(short)wa[i >> 2][i & 3];
    w1[i] = (_Float16)(short)wa[2 + (i >> 2)][i & 3];
  }
  *(v8h*)(dA)     = a0;
  *(v8h*)(dA + 8) = a1;
  *(v8h*)(dW)     = w0;
  *(v8h*)(dW + 8) = w1;
}

__global__ __launch_bounds__(512)
void qlinear_wmma_f16(const float* __restrict__ x,
                      const int*   __restrict__ w8,
                      const float* __restrict__ scale,
                      const float* __restrict__ zero,
                      const float* __restrict__ bias,
                      float* __restrict__ out) {
  extern __shared__ __align__(16) _Float16 smem[];
  _Float16* sA = smem;               // [2][TILEH] x tile,  [row][k] f16
  _Float16* sB = smem + 2 * TILEH;   // [2][TILEH] W tile,  [n][k]  f16
  float*  sRow = (float*)(smem + 4 * TILEH);   // [BM] rowsum(x)

  const int tid  = threadIdx.x;
  const int lane = tid & 31;
  const int wave = tid >> 5;       // 0..15
  const int wm   = wave >> 2;      // 0..3  (M direction, 64 rows each)
  const int wn   = wave & 3;       // 0..3  (N direction, 64 cols each)
  const int r    = lane & 15;
  const int hi   = lane >> 4;      // half-wave select

  const int mb = blockIdx.y * BM;
  const int nb = blockIdx.x * BN;

  // staging: thread t loads 16 contiguous elements of one row (row = t>>1)
  const int srow = tid >> 1;            // 0..255
  const int skh  = (tid & 1) * 16;      // 0 or 16 within BK
  const float* gA = x  + (size_t)(mb + srow) * KDIM + skh;
  const int*   gW = w8 + (size_t)(nb + srow) * KDIM + skh;

  const v8f z8 = {0.f, 0.f, 0.f, 0.f, 0.f, 0.f, 0.f, 0.f};
  v8f acc[4][4];
#pragma unroll
  for (int mt = 0; mt < 4; ++mt)
#pragma unroll
    for (int nt = 0; nt < 4; ++nt) acc[mt][nt] = z8;

  // per-staging-thread running rowsum of x (fp32, published via LDS atomic)
  float racc = 0.f;
  if (tid < BM) sRow[tid] = 0.f;

  // ---- prologue: stage K-tile 0 into LDS buffer 0 ----
  v4f fa[4];
  v4i wa[4];
  load_stage(gA, gW, fa, wa);
  {
    v4f s = (fa[0] + fa[1]) + (fa[2] + fa[3]);
    racc += (s[0] + s[1]) + (s[2] + s[3]);
  }
  store_stage(&sA[srow * SAS + skh], &sB[srow * SAS + skh], fa, wa);
  __syncthreads();

  for (int kt = 0; kt < NKT; ++kt) {
    const int cur = kt & 1;

    // issue next tile's global loads early (overlap with WMMA compute)
    if (kt + 1 < NKT) {
      load_stage(gA + (size_t)(kt + 1) * BK, gW + (size_t)(kt + 1) * BK, fa, wa);
      v4f s = (fa[0] + fa[1]) + (fa[2] + fa[3]);
      racc += (s[0] + s[1]) + (s[2] + s[3]);
    }
    if (kt + 2 < NKT) {
      __builtin_prefetch(gA + (size_t)(kt + 2) * BK, 0, 3);   // global_prefetch_b8
      __builtin_prefetch(gW + (size_t)(kt + 2) * BK, 0, 3);
    }

    // ---- fragment loads from LDS (layouts per CDNA5 ISA 7.12.2) ----
    // A 16x32 f16: lane holds row (m0+r); elems 0-7 = K(hi*8+0..7), 8-15 = K(16+hi*8+0..7)
    const _Float16* a0 = sA + cur * TILEH;
    const _Float16* b0 = sB + cur * TILEH;
    v16h af[4];
#pragma unroll
    for (int mt = 0; mt < 4; ++mt) {
      const _Float16* pa = a0 + (wm * 64 + mt * 16 + r) * SAS + hi * 8;
      v8h lo = *(const v8h*)(pa);
      v8h hh = *(const v8h*)(pa + 16);
      af[mt] = __builtin_shufflevector(lo, hh, 0, 1, 2, 3, 4, 5, 6, 7,
                                               8, 9, 10, 11, 12, 13, 14, 15);
    }
    // B 32x16 f16 streamed one nt at a time to bound register liveness.
    // lane holds col (n0+r); elems 0-7 = K(hi*8+0..7), 8-15 = K(16+hi*8+0..7)
#pragma unroll
    for (int nt = 0; nt < 4; ++nt) {
      const _Float16* pb = b0 + (wn * 64 + nt * 16 + r) * SAS + hi * 8;
      v8h lo = *(const v8h*)(pb);
      v8h hh = *(const v8h*)(pb + 16);
      v16h bf = __builtin_shufflevector(lo, hh, 0, 1, 2, 3, 4, 5, 6, 7,
                                                8, 9, 10, 11, 12, 13, 14, 15);
#pragma unroll
      for (int mt = 0; mt < 4; ++mt)
        acc[mt][nt] = __builtin_amdgcn_wmma_f32_16x16x32_f16(
            false, af[mt], false, bf, (short)0, acc[mt][nt], false, false);
    }

    // stage next tile into the buffer nobody reads this iteration; ONE
    // barrier per K-step is sufficient with disjoint double buffers.
    if (kt + 1 < NKT)
      store_stage(&sA[(cur ^ 1) * TILEH + srow * SAS + skh],
                  &sB[(cur ^ 1) * TILEH + srow * SAS + skh], fa, wa);
    __syncthreads();
  }

  // ---- publish rowsums (ds_add_f32; 2 staging threads per row) ----
  atomicAdd(&sRow[srow], racc);
  __syncthreads();

  // ---- epilogue: y = scale*acc - scale*zero*rowsum + bias ----
  // C/D layout: lane holds col n0+(lane&15); VGPR g holds row m0 + hi*8 + g.
  // Output is write-once: non-temporal stores keep x/W tiles resident in L2.
#pragma unroll
  for (int nt = 0; nt < 4; ++nt) {
    const int n  = nb + wn * 64 + nt * 16 + r;
    const float sc   = scale[n];
    const float coef = -sc * zero[n];
    const float bi   = bias[n];
#pragma unroll
    for (int mt = 0; mt < 4; ++mt) {
      const int lrow = wm * 64 + mt * 16 + hi * 8;
      const int grow = mb + lrow;
#pragma unroll
      for (int g = 0; g < 8; ++g) {
        const float rv  = sRow[lrow + g];
        const float val = sc * acc[mt][nt][g] + coef * rv + bi;
        __builtin_nontemporal_store(val, &out[(size_t)(grow + g) * NDIM + n]);
      }
    }
  }
}

extern "C" void kernel_launch(void* const* d_in, const int* in_sizes, int n_in,
                              void* d_out, int out_size, void* d_ws, size_t ws_size,
                              hipStream_t stream) {
  (void)in_sizes; (void)n_in; (void)out_size; (void)d_ws; (void)ws_size;
  const float* x     = (const float*)d_in[0];
  const int*   w8    = (const int*)d_in[1];
  const float* scale = (const float*)d_in[2];
  const float* zero  = (const float*)d_in[3];
  const float* bias  = (const float*)d_in[4];
  float* out = (float*)d_out;

  dim3 grid(NDIM / BN, MDIM / BM);   // (43, 32)
  dim3 block(512);                   // 16 wave32s, 4x4 wave grid per 256x256 tile
  qlinear_wmma_f16<<<grid, block, LDS_BYTES, stream>>>(x, w8, scale, zero, bias, out);
}
